// OutputBlock_80006650790312
// MI455X (gfx1250) — matrix operand order repaired and checked by reading, not compiled, not run on hardware
//
#include <hip/hip_runtime.h>
#include <hip/hip_bf16.h>

typedef __attribute__((ext_vector_type(16))) __bf16 v16bf;
typedef __attribute__((ext_vector_type(8)))  float  v8f;
typedef __attribute__((ext_vector_type(4)))  unsigned int u32x4;
typedef __attribute__((ext_vector_type(4)))  int          i32x4;

#define NNODES  100000
#define EDGES   1000000
#define HIDDEN  128
#define OUT_EMB 512
#define OUT_CH  128

#define AS_GLOBAL __attribute__((address_space(1)))
#define AS_LDS    __attribute__((address_space(3)))

// ---- gfx1250 async global->LDS path (guarded; fallback = sync reg staging) ----
#if defined(__has_builtin)
#  if __has_builtin(__builtin_amdgcn_global_load_async_to_lds_b128)
#    define HAS_ASYNC_COPY 1
#  endif
#endif
#ifndef HAS_ASYNC_COPY
#  define HAS_ASYNC_COPY 0
#endif
#if HAS_ASYNC_COPY && defined(__has_builtin)
#  if __has_builtin(__builtin_amdgcn_s_wait_asynccnt)
#    define HAS_WAIT_BUILTIN 1
#  endif
#endif
#ifndef HAS_WAIT_BUILTIN
#  define HAS_WAIT_BUILTIN 0
#endif

__device__ __forceinline__ void async_copy_b128(const __bf16* g, __bf16* l) {
#if HAS_ASYNC_COPY
    __builtin_amdgcn_global_load_async_to_lds_b128(
        (AS_GLOBAL i32x4*)g, (AS_LDS i32x4*)l, 0, 0);
#else
    *(u32x4*)l = *(const u32x4*)g;
#endif
}

template <int N>
__device__ __forceinline__ void wait_async() {
#if HAS_ASYNC_COPY
#  if HAS_WAIT_BUILTIN
    __builtin_amdgcn_s_wait_asynccnt(N);
#  else
    asm volatile("s_wait_asynccnt %0" :: "i"(N) : "memory");
#  endif
#endif
}

// ---------------------------------------------------------------------------
// fp32 -> bf16 weight conversion
// ---------------------------------------------------------------------------
__global__ void cvt_bf16(const float* __restrict__ s, __bf16* __restrict__ d, int n) {
    int i = blockIdx.x * blockDim.x + threadIdx.x;
    if (i < n) d[i] = (__bf16)s[i];
}

// ---------------------------------------------------------------------------
// Edge kernel: e = (rbf @ W_rbf.T) * x, scatter-add into h0[node].
// One wave (32 lanes) per edge; lane handles 4 channels (128 total).
// h0 (51 MB) is L2-resident (192 MB L2), atomics resolve at L2.
// ---------------------------------------------------------------------------
__global__ __launch_bounds__(256)
void edge_scatter(const float* __restrict__ rbf, const float* __restrict__ x,
                  const int* __restrict__ idx, const float* __restrict__ Wrbf,
                  float* __restrict__ h0, int E) {
    long tid  = (long)blockIdx.x * blockDim.x + threadIdx.x;
    int  e    = (int)(tid >> 5);
    int  lane = (int)(tid & 31);
    if (e >= E) return;

    const float r0 = rbf[e * 6 + 0], r1 = rbf[e * 6 + 1], r2 = rbf[e * 6 + 2];
    const float r3 = rbf[e * 6 + 3], r4 = rbf[e * 6 + 4], r5 = rbf[e * 6 + 5];
    const int node = idx[e];

    const float4 xv = *(const float4*)(x + (size_t)e * HIDDEN + lane * 4);
    float* dst = h0 + (size_t)node * HIDDEN + lane * 4;

    const float xs[4] = {xv.x, xv.y, xv.z, xv.w};
#pragma unroll
    for (int j = 0; j < 4; ++j) {
        const int c = lane * 4 + j;
        const float* w = Wrbf + c * 6;
        float p = r0 * w[0] + r1 * w[1] + r2 * w[2] + r3 * w[3] + r4 * w[4] + r5 * w[5];
        atomicAdd(dst + j, p * xs[j]);
    }
}

// ---------------------------------------------------------------------------
// Double-buffered bf16 WMMA GEMM:  C[M,N] = act( A[M,K] @ Bw[N,K]^T + bias )
// Block tile 64(M) x 128(N), BK = 64 (two v_wmma_f32_16x16x32_bf16 K-chunks
// per LDS stage -> 8 WMMAs per barrier pair). 8 waves: 2(M) x 4(N), each wave
// owns a 32x32 tile (2x2 fp32 accumulators).
// Stage s+1 is fetched via GLOBAL_LOAD_ASYNC_TO_LDS (ASYNCcnt) while stage s
// feeds the matrix pipes; in-order async completion means waiting
// asynccnt <= P_OPS after issuing stage s+1 guarantees stage s has landed.
// ---------------------------------------------------------------------------
template <bool A_F32, bool SILU, bool OUT_F32>
__global__ __launch_bounds__(256)
void gemm_wmma(const void* __restrict__ Aptr, const __bf16* __restrict__ Bw,
               const float* __restrict__ bias, void* __restrict__ Cptr,
               int M, int N, int K) {
    constexpr int BM = 64, BN = 128, BK = 64;
    constexpr int P_OPS = A_F32 ? 4 : 6;   // async ops issued per thread per stage
    __shared__ alignas(16) __bf16 As[2][BM][BK];
    __shared__ alignas(16) __bf16 Bs[2][BN][BK];

    const int tid  = threadIdx.x;
    const int lane = tid & 31;
    const int wave = tid >> 5;      // 0..7
    const int wm   = wave & 1;      // 2 waves along M
    const int wn   = wave >> 1;     // 4 waves along N
    const int m0   = blockIdx.y * BM;
    const int n0   = blockIdx.x * BN;

    const int halo = lane >> 4;     // lane half (ISA fragment layout)
    const int l15  = lane & 15;

    // staging map: A = 16 elems/thread, B = 32 elems/thread
    const int ar = tid >> 2, ak = (tid & 3) * 16;
    const int br = tid >> 1, bk = (tid & 1) * 32;
    int gmA = m0 + ar; if (gmA >= M) gmA = M - 1;   // clamp (stores guarded)

    v8f acc[2][2] = {};

    auto prefetch = [&](int buf, int k0) {
        if constexpr (A_F32) {
            // fp32 input: sync load + convert (async path can't convert)
            const float* A   = (const float*)Aptr;
            const float* src = A + (size_t)gmA * K + k0 + ak;
            alignas(16) __bf16 t[16];
#pragma unroll
            for (int j = 0; j < 16; j += 4) {
                const float4 f = *(const float4*)(src + j);
                t[j + 0] = (__bf16)f.x; t[j + 1] = (__bf16)f.y;
                t[j + 2] = (__bf16)f.z; t[j + 3] = (__bf16)f.w;
            }
            *(u32x4*)&As[buf][ar][ak]     = *(const u32x4*)&t[0];
            *(u32x4*)&As[buf][ar][ak + 8] = *(const u32x4*)&t[8];
        } else {
            const __bf16* A   = (const __bf16*)Aptr;
            const __bf16* src = A + (size_t)gmA * K + k0 + ak;
            async_copy_b128(src,     &As[buf][ar][ak]);
            async_copy_b128(src + 8, &As[buf][ar][ak + 8]);
        }
        const __bf16* bsrc = Bw + (size_t)(n0 + br) * K + k0 + bk;
        async_copy_b128(bsrc,      &Bs[buf][br][bk]);
        async_copy_b128(bsrc + 8,  &Bs[buf][br][bk + 8]);
        async_copy_b128(bsrc + 16, &Bs[buf][br][bk + 16]);
        async_copy_b128(bsrc + 24, &Bs[buf][br][bk + 24]);
    };

    const int nk = K / BK;
    prefetch(0, 0);
    for (int s = 0; s < nk; ++s) {
        if (s + 1 < nk) {
            prefetch((s + 1) & 1, (s + 1) * BK);
            wait_async<P_OPS>();          // stage s landed; stage s+1 in flight
        } else {
            wait_async<0>();
        }
        __syncthreads();

        const int buf = s & 1;
#pragma unroll
        for (int kc = 0; kc < 2; ++kc) {
            v16bf afrag[2], bfrag[2];
#pragma unroll
            for (int mt = 0; mt < 2; ++mt) {
                const int row = wm * 32 + mt * 16 + l15;
                const int ka  = kc * 32 + halo * 8;   // K 0-7/16-23 vs 8-15/24-31
                union { u32x4 u[2]; v16bf v; } ua;
                ua.u[0] = *(const u32x4*)&As[buf][row][ka];
                ua.u[1] = *(const u32x4*)&As[buf][row][ka + 16];
                afrag[mt] = ua.v;
            }
#pragma unroll
            for (int nt = 0; nt < 2; ++nt) {
                const int col = wn * 32 + nt * 16 + l15;
                const int kb  = kc * 32 + halo * 8;   // K 0-7/16-23 vs 8-15/24-31
                union { u32x4 u[2]; v16bf v; } ub;
                ub.u[0] = *(const u32x4*)&Bs[buf][col][kb];
                ub.u[1] = *(const u32x4*)&Bs[buf][col][kb + 16];
                bfrag[nt] = ub.v;
            }
#pragma unroll
            for (int mt = 0; mt < 2; ++mt)
#pragma unroll
                for (int nt = 0; nt < 2; ++nt)
                    acc[mt][nt] = __builtin_amdgcn_wmma_f32_16x16x32_bf16(
                        false, afrag[mt], false, bfrag[nt],
                        (short)0, acc[mt][nt], false, false);
        }
        __syncthreads();   // protect buf from the next stage's prefetch
    }

    // ---- epilogue: bias, SiLU, store (fp32 or bf16) ----
#pragma unroll
    for (int mt = 0; mt < 2; ++mt) {
#pragma unroll
        for (int nt = 0; nt < 2; ++nt) {
            const int col = n0 + wn * 32 + nt * 16 + l15;
            const float b = bias ? bias[col] : 0.0f;
#pragma unroll
            for (int v = 0; v < 8; ++v) {
                const int row = m0 + wm * 32 + mt * 16 + halo * 8 + v;  // ISA C/D layout
                if (row < M) {
                    float c = acc[mt][nt][v] + b;
                    if (SILU) c = c / (1.0f + __expf(-c));
                    if (OUT_F32) ((float*)Cptr)[(size_t)row * N + col] = c;
                    else ((__bf16*)Cptr)[(size_t)row * N + col] = (__bf16)c;
                }
            }
        }
    }
}

// ---------------------------------------------------------------------------
extern "C" void kernel_launch(void* const* d_in, const int* in_sizes, int n_in,
                              void* d_out, int out_size, void* d_ws, size_t ws_size,
                              hipStream_t stream) {
    const float* x    = (const float*)d_in[0];
    const float* rbf  = (const float*)d_in[1];
    const int*   idx  = (const int*)d_in[2];
    // d_in[3] = num_nodes scalar (compile-time constant here)
    const float* Wrbf = (const float*)d_in[4];
    const float* Wup  = (const float*)d_in[5];
    const float* Ws   = (const float*)d_in[6];
    const float* bs   = (const float*)d_in[7];
    const float* Wout = (const float*)d_in[8];

    // workspace carve-out (256B aligned slices)
    char*  ws  = (char*)d_ws;
    size_t off = 0;
    auto alloc = [&](size_t bytes) -> char* {
        char* p = ws + off;
        off = (off + bytes + 255) & ~(size_t)255;
        return p;
    };
    float*  h0    = (float*) alloc((size_t)NNODES * HIDDEN * sizeof(float));
    __bf16* WupB  = (__bf16*)alloc((size_t)OUT_EMB * HIDDEN * 2);
    __bf16* WsB   = (__bf16*)alloc((size_t)3 * OUT_EMB * OUT_EMB * 2);
    __bf16* WoutB = (__bf16*)alloc((size_t)OUT_CH * OUT_EMB * 2);
    __bf16* actA  = (__bf16*)alloc((size_t)NNODES * OUT_EMB * 2);
    __bf16* actB  = (__bf16*)alloc((size_t)NNODES * OUT_EMB * 2);

    (void)hipMemsetAsync(h0, 0, (size_t)NNODES * HIDDEN * sizeof(float), stream);

    { int n = OUT_EMB * HIDDEN;      cvt_bf16<<<(n + 255) / 256, 256, 0, stream>>>(Wup,  WupB,  n); }
    { int n = 3 * OUT_EMB * OUT_EMB; cvt_bf16<<<(n + 255) / 256, 256, 0, stream>>>(Ws,   WsB,   n); }
    { int n = OUT_CH * OUT_EMB;      cvt_bf16<<<(n + 255) / 256, 256, 0, stream>>>(Wout, WoutB, n); }

    {   // fused rbf-proj * x + scatter-sum
        long t = (long)EDGES * 32;
        edge_scatter<<<(int)((t + 255) / 256), 256, 0, stream>>>(rbf, x, idx, Wrbf, h0, EDGES);
    }

    const int gy = (NNODES + 63) / 64;
    // lin_up: [100k,128] x [512,128]^T -> bf16 activations
    gemm_wmma<true, false, false><<<dim3(OUT_EMB / 128, gy), 256, 0, stream>>>(
        h0, WupB, nullptr, actA, NNODES, OUT_EMB, HIDDEN);

    // 3x SiLU MLP layers, bf16 ping-pong
    const __bf16* cur = actA;
    __bf16*       nxt = actB;
    for (int l = 0; l < 3; ++l) {
        gemm_wmma<false, true, false><<<dim3(OUT_EMB / 128, gy), 256, 0, stream>>>(
            (const void*)cur, WsB + (size_t)l * OUT_EMB * OUT_EMB,
            bs + (size_t)l * OUT_EMB, nxt, NNODES, OUT_EMB, OUT_EMB);
        __bf16* t = (__bf16*)cur; cur = nxt; nxt = t;
    }

    // final projection -> fp32 d_out
    gemm_wmma<false, false, true><<<dim3(OUT_CH / 128, gy), 256, 0, stream>>>(
        (const void*)cur, WoutB, nullptr, d_out, NNODES, OUT_CH, OUT_EMB);
}
